// Dot_product_attention_86045374808389
// MI455X (gfx1250) — compile-verified, matrix operand order checked by reading, not run on hardware
//
#include <hip/hip_runtime.h>

#define TT 2048
#define FF 128
#define NH 16

typedef __attribute__((ext_vector_type(16))) __bf16 v16bf;
typedef __attribute__((ext_vector_type(8)))  float  v8f;

union BFrag {
  v16bf v;
  unsigned short u[16];
  uint4 q[2];
};

// Hardware f32 -> bf16 (fptrunc -> native cvt on gfx1250).
__device__ __forceinline__ unsigned short bf(float f) {
  __bf16 h = (__bf16)f;
  return __builtin_bit_cast(unsigned short, h);
}

__device__ __forceinline__ unsigned pack2(float lo, float hi) {
  return (unsigned)bf(lo) | ((unsigned)bf(hi) << 16);
}

// 16-lane max butterfly via ds_swizzle (imm XOR pattern, no addr VGPR).
__device__ __forceinline__ float red_max16(float x) {
  x = fmaxf(x, __builtin_bit_cast(float, __builtin_amdgcn_ds_swizzle(
                   __builtin_bit_cast(int, x), 0x041f)));  // xor 1
  x = fmaxf(x, __builtin_bit_cast(float, __builtin_amdgcn_ds_swizzle(
                   __builtin_bit_cast(int, x), 0x081f)));  // xor 2
  x = fmaxf(x, __builtin_bit_cast(float, __builtin_amdgcn_ds_swizzle(
                   __builtin_bit_cast(int, x), 0x101f)));  // xor 4
  x = fmaxf(x, __builtin_bit_cast(float, __builtin_amdgcn_ds_swizzle(
                   __builtin_bit_cast(int, x), 0x201f)));  // xor 8
  return x;
}

// Broadcast from lane (lane & 0x10): src = (lane & and_mask); and=0x10, or=0, xor=0.
__device__ __forceinline__ float bcast16(float x) {
  return __builtin_bit_cast(float, __builtin_amdgcn_ds_swizzle(
             __builtin_bit_cast(int, x), 0x0010));
}

// Q A-fragment load from global: elements [0..7] and [16..23] from p.
__device__ __forceinline__ v16bf ld_qfrag(const float* __restrict__ p, float scale) {
  const float4* p4 = reinterpret_cast<const float4*>(p);
  float t[16];
  *reinterpret_cast<float4*>(&t[0])  = p4[0];
  *reinterpret_cast<float4*>(&t[4])  = p4[1];
  *reinterpret_cast<float4*>(&t[8])  = p4[4];
  *reinterpret_cast<float4*>(&t[12]) = p4[5];
  BFrag fr;
#pragma unroll
  for (int e = 0; e < 16; ++e) fr.u[e] = bf(t[e] * scale);
  return fr.v;
}

__device__ __forceinline__ v8f wmma_bf16(v16bf a, v16bf b, v8f c) {
  return __builtin_amdgcn_wmma_f32_16x16x32_bf16(false, a, false, b, (short)0, c,
                                                 false, false);
}

__global__ __launch_bounds__(256)
void fa_fwd_kernel(const float* __restrict__ qg, const float* __restrict__ kg,
                   const float* __restrict__ vg, const int* __restrict__ maskedp,
                   float* __restrict__ outg) {
  __shared__ __attribute__((aligned(16))) unsigned short Klds[32 * 128];     // 8 KB
  __shared__ __attribute__((aligned(16))) unsigned short Vlds[128 * 32];     // 8 KB
  __shared__ __attribute__((aligned(16))) unsigned short Plds[8 * 16 * 32];  // 8 KB

  const int tid  = threadIdx.x;
  const int lane = tid & 31;
  const int wave = tid >> 5;
  const int n    = lane & 15;
  const int half = lane >> 4;
  const int head = blockIdx.x >> 4;
  const int tg   = blockIdx.x & 15;
  const int brow0 = tg * 128;
  const int row0  = brow0 + wave * 16;
  const int masked = maskedp[0];

  const size_t base = (size_t)head * TT * FF;
  // (1/sqrt(128)) * log2(e): softmax runs in exp2 domain
  const float scale = 0.12753132f;

  const float* qrow = qg + base + (size_t)(row0 + n) * FF + 8 * half;
  v16bf qa[4];
#pragma unroll
  for (int w = 0; w < 4; ++w) qa[w] = ld_qfrag(qrow + 32 * w, scale);

  // constant "ones column" B-fragment: V_ones[s][c] = (c==0) ? 1 : 0
  BFrag vones;
#pragma unroll
  for (int e = 0; e < 16; ++e) vones.u[e] = (n == 0) ? (unsigned short)0x3F80 : 0;

  v8f o[8];
#pragma unroll
  for (int jf = 0; jf < 8; ++jf) o[jf] = (v8f){0, 0, 0, 0, 0, 0, 0, 0};
  v8f ol = (v8f){0, 0, 0, 0, 0, 0, 0, 0};  // row-sum accumulator (col 0 of lanes n==0)
  float mrun[8];
#pragma unroll
  for (int r = 0; r < 8; ++r) mrun[r] = -1e30f;

  const int sEndW = masked ? (row0 + 16) : TT;
  const int sEndB = masked ? (brow0 + 128) : TT;
  const float* kbase = kg + base;
  const float* vbase = vg + base;
  unsigned short* myP = &Plds[wave * 512];

  for (int s0 = 0; s0 < sEndB; s0 += 32) {
    __syncthreads();

    // ===== cooperative stage: 32x128 K (row-major) and V (transposed) -> LDS
#pragma unroll
    for (int ii = 0; ii < 4; ++ii) {
      const int idx  = tid * 4 + ii;
      const int elem = idx * 4;
      const int s    = elem >> 7;
      const int f    = elem & 127;
      const float4 k4 = *reinterpret_cast<const float4*>(
          kbase + (size_t)(s0 + s) * FF + f);
      const float4 v4 = *reinterpret_cast<const float4*>(
          vbase + (size_t)(s0 + s) * FF + f);
      unsigned* kd = reinterpret_cast<unsigned*>(&Klds[s * 128 + f]);
      kd[0] = pack2(k4.x, k4.y);
      kd[1] = pack2(k4.z, k4.w);
      Vlds[(f + 0) * 32 + s] = bf(v4.x);
      Vlds[(f + 1) * 32 + s] = bf(v4.y);
      Vlds[(f + 2) * 32 + s] = bf(v4.z);
      Vlds[(f + 3) * 32 + s] = bf(v4.w);
      if (s0 + 32 < sEndB) {
        __builtin_prefetch(kbase + (size_t)(s0 + 32 + s) * FF + f, 0, 1);
        __builtin_prefetch(vbase + (size_t)(s0 + 32 + s) * FF + f, 0, 1);
      }
    }
    __syncthreads();

    if (s0 >= sEndW) continue;  // past causal limit: barriers only

    // ===== GEMM1: S = Q * K^T (exp2 domain, scale folded into Q)
    v8f c0 = (v8f){0, 0, 0, 0, 0, 0, 0, 0};
    v8f c1 = c0;
    {
      const unsigned short* kr0 = &Klds[n * 128 + 8 * half];
      const unsigned short* kr1 = &Klds[(16 + n) * 128 + 8 * half];
#pragma unroll
      for (int w = 0; w < 4; ++w) {
        BFrag kb;
        const uint4* p = reinterpret_cast<const uint4*>(kr0 + 32 * w);
        kb.q[0] = p[0];
        kb.q[1] = p[2];
        c0 = wmma_bf16(qa[w], kb.v, c0);
      }
#pragma unroll
      for (int w = 0; w < 4; ++w) {
        BFrag kb;
        const uint4* p = reinterpret_cast<const uint4*>(kr1 + 32 * w);
        kb.q[0] = p[0];
        kb.q[1] = p[2];
        c1 = wmma_bf16(qa[w], kb.v, c1);
      }
    }

    // ===== online softmax, exp2 domain (row-sum folded into GEMM2)
    const bool needMask = (masked != 0) && (s0 + 31 > row0);
#pragma unroll
    for (int r = 0; r < 8; ++r) {
      const int row = row0 + r + 8 * half;
      float v0 = c0[r], v1 = c1[r];
      if (needMask) {
        if (s0 + n > row)      v0 = -1e30f;
        if (s0 + 16 + n > row) v1 = -1e30f;
      }
      const float rmax  = red_max16(fmaxf(v0, v1));
      const float nm    = fmaxf(mrun[r], rmax);
      const float alpha = exp2f(mrun[r] - nm);
      mrun[r] = nm;
      const float p0 = exp2f(v0 - nm);
      const float p1 = exp2f(v1 - nm);
      ol[r] *= alpha;
#pragma unroll
      for (int jf = 0; jf < 8; ++jf) o[jf][r] *= alpha;
      const int prow = r + 8 * half;
      myP[prow * 32 + n]      = bf(p0);
      myP[prow * 32 + 16 + n] = bf(p1);
    }

    asm volatile("s_wait_dscnt 0x0" ::: "memory");

    BFrag pa;  // P as A-fragment
    {
      const uint4* ps = reinterpret_cast<const uint4*>(&myP[n * 32 + 8 * half]);
      pa.q[0] = ps[0];
      pa.q[1] = ps[2];
    }

    // ===== GEMM2: O += P * V ; row-sums: ol += P * ones
#pragma unroll
    for (int jf = 0; jf < 8; ++jf) {
      BFrag vb;
      const uint4* p =
          reinterpret_cast<const uint4*>(&Vlds[(16 * jf + n) * 32 + 8 * half]);
      vb.q[0] = p[0];
      vb.q[1] = p[2];
      o[jf] = wmma_bf16(pa.v, vb.v, o[jf]);
    }
    ol = wmma_bf16(pa.v, vones.v, ol);
  }

  // ===== epilogue: l = bcast(ol col0), O *= 1/l
  float* ob = outg + base;
  float rl[8];
#pragma unroll
  for (int r = 0; r < 8; ++r) rl[r] = __builtin_amdgcn_rcpf(bcast16(ol[r]));
#pragma unroll
  for (int jf = 0; jf < 8; ++jf) {
#pragma unroll
    for (int r = 0; r < 8; ++r) {
      const int row = row0 + r + 8 * half;
      ob[(size_t)row * FF + 16 * jf + n] = o[jf][r] * rl[r];
    }
  }
}

extern "C" void kernel_launch(void* const* d_in, const int* in_sizes, int n_in,
                              void* d_out, int out_size, void* d_ws, size_t ws_size,
                              hipStream_t stream) {
  const float* q = (const float*)d_in[0];
  const float* k = (const float*)d_in[1];
  const float* v = (const float*)d_in[2];
  const int* masked = (const int*)d_in[3];
  float* out = (float*)d_out;

  dim3 grid(NH * (TT / 128));  // 256 blocks
  dim3 block(256);             // 8 waves * 16-row Q tiles
  hipLaunchKernelGGL(fa_fwd_kernel, grid, block, 0, stream, q, k, v, masked, out);
}
